// MOA_7490422964585
// MI455X (gfx1250) — matrix operand order repaired and checked by reading, compiled, not run on hardware
//
#include <hip/hip_runtime.h>
#include <hip/hip_bf16.h>

typedef __attribute__((ext_vector_type(16))) __bf16 v16bf;
typedef __attribute__((ext_vector_type(8)))  float  v8f;

static constexpr int B_ = 4, N_ = 4096, D_ = 2048, E_ = 8, R_ = 128, K_ = 1024;

__device__ __forceinline__ unsigned f2bf(float f) {
  unsigned u = __float_as_uint(f);
  return (u + 0x7FFFu + ((u >> 16) & 1u)) >> 16;   // RNE fp32 -> bf16
}
__device__ __forceinline__ unsigned packbf(float lo, float hi) {
  return f2bf(lo) | (f2bf(hi) << 16);
}
union Frag { unsigned u[8]; uint4 q[2]; v16bf v; };

// ---------------------------------------------------------------------------
// Kernel 1: gate logits  [B,N,E] = x @ Wg + bg   (one wave32 per token)
// ---------------------------------------------------------------------------
__global__ __launch_bounds__(256) void moa_gate_kernel(
    const float* __restrict__ x, const float* __restrict__ Wg,
    const float* __restrict__ bg, float* __restrict__ logits)
{
  const int t    = blockIdx.x * 8 + (threadIdx.x >> 5);   // token in [0, B*N)
  const int lane = threadIdx.x & 31;
  const float* xr = x + (size_t)t * D_;
  float a[E_];
#pragma unroll
  for (int e = 0; e < E_; ++e) a[e] = 0.0f;
  for (int d = lane; d < D_; d += 32) {
    float xv = xr[d];
    const float* wr = Wg + (size_t)d * E_;
#pragma unroll
    for (int e = 0; e < E_; ++e) a[e] = fmaf(xv, wr[e], a[e]);
  }
#pragma unroll
  for (int off = 16; off > 0; off >>= 1) {
#pragma unroll
    for (int e = 0; e < E_; ++e) a[e] += __shfl_xor(a[e], off, 32);
  }
  if (lane < E_) logits[(size_t)t * E_ + lane] = a[lane] + bg[lane];
}

// ---------------------------------------------------------------------------
// Kernel 2: per-(b,e) top-K=1024 of N=4096 logits via 4-pass radix select
// ---------------------------------------------------------------------------
__global__ __launch_bounds__(256) void moa_topk_kernel(
    const float* __restrict__ logits, int* __restrict__ idxOut,
    float* __restrict__ gateOut)
{
  const int e = blockIdx.x, b = blockIdx.y, tid = threadIdx.x;
  __shared__ unsigned keys[N_];
  __shared__ unsigned hist[256];
  __shared__ int sSel, sRem, cntG, cntE;

  for (int n = tid; n < N_; n += 256) {
    unsigned u = __float_as_uint(logits[((size_t)b * N_ + n) * E_ + e]);
    keys[n] = (u & 0x80000000u) ? ~u : (u | 0x80000000u);  // order-preserving
  }
  if (tid == 0) { sRem = K_; cntG = 0; cntE = 0; }
  __syncthreads();

  unsigned prefix = 0, pmask = 0;
  for (int pass = 0; pass < 4; ++pass) {
    const int shift = 24 - 8 * pass;
    hist[tid] = 0;
    __syncthreads();
    for (int n = tid; n < N_; n += 256) {
      unsigned k = keys[n];
      if ((k & pmask) == prefix) atomicAdd(&hist[(k >> shift) & 255u], 1u);
    }
    __syncthreads();
    if (tid == 0) {
      int rem = sRem; unsigned cum = 0; int sel = 0;
      for (int d = 255; d >= 0; --d) {
        unsigned c = hist[d];
        if (cum + c >= (unsigned)rem) { sel = d; sRem = rem - (int)cum; break; }
        cum += c;
      }
      sSel = sel;
    }
    __syncthreads();
    prefix |= ((unsigned)sSel) << shift;
    pmask  |= 0xFFu << shift;
    __syncthreads();
  }

  const unsigned T = prefix;           // exact K-th largest key
  const int rem   = sRem;              // #ties at T we must keep
  const int numG  = K_ - rem;
  const size_t base = ((size_t)(b * E_) + e) * K_;
  for (int n = tid; n < N_; n += 256) {
    unsigned k = keys[n];
    int pos = -1;
    if (k > T) {
      pos = atomicAdd(&cntG, 1);
    } else if (k == T) {
      int p = atomicAdd(&cntE, 1);
      if (p < rem) pos = numG + p;
    }
    if (pos >= 0) {
      unsigned fb = (k & 0x80000000u) ? (k ^ 0x80000000u) : ~k;
      idxOut[base + pos]  = n;
      gateOut[base + pos] = __uint_as_float(fb);
    }
  }
}

// ---------------------------------------------------------------------------
// Kernel 3: fused gather -> down GEMM -> relu -> up GEMM -> gated scatter-add
// One block per (b, e, 128-token tile); 8 waves; bf16 WMMA 16x16x32.
// All LDS fragments are contiguous 16B quads (ds_load_b128) with padded
// strides chosen so 16-lane b128 reads are bank-conflict free.
// ---------------------------------------------------------------------------
__global__ __launch_bounds__(256) void moa_adapter_kernel(
    const float* __restrict__ x,  const float* __restrict__ Wd,
    const float* __restrict__ bd, const float* __restrict__ Wu,
    const float* __restrict__ bu, const int* __restrict__ idx,
    const float* __restrict__ gates, float* __restrict__ out)
{
  const int mt = blockIdx.x, e = blockIdx.y, b = blockIdx.z;
  const int tid  = threadIdx.x;
  const int lane = tid & 31, wave = tid >> 5;
  const int lh = (lane < 16) ? 0 : 1;      // lane half (K split per ISA layout)
  const int lc = lane & 15;                // column / row-within-16

  // Padded strides: 20 dw (20*r mod 64 covers all banks), 68 dw (4*c mod 64).
  __shared__ union {
    struct {
      unsigned a[128][20];     // gathered X tile, [token][k-pair], pad 16->20
      unsigned bwT[128][20];   // Wd tile transposed, [r-col][k-pair]
    } p1;
    unsigned b2T[64][68];      // Wu tile transposed, [d-col][k-pair], pad 64->68
  } sm;
  __shared__ unsigned short hLDS[128][136];  // bf16 h, padded row (272B, 16B-mult)
  __shared__ int   tokLds[128];
  __shared__ float gateLds[128];

  if (tid < 128) {
    int g = ((b * E_) + e) * K_ + mt * 128 + tid;
    tokLds[tid]  = idx[g];
    gateLds[tid] = gates[g];
  }
  __syncthreads();

  // ----- Phase 1: h[128 x 128] = relu( X_in[128 x D] @ Wd[D x R] + bd ) -----
  const v8f zf = {0.f,0.f,0.f,0.f,0.f,0.f,0.f,0.f};
  v8f acc[8];
#pragma unroll
  for (int i = 0; i < 8; ++i) acc[i] = zf;

  const int rowS = tid >> 1, halfS = tid & 1;                 // staging coords
  const float* xrow = x + ((size_t)(b * N_) + tokLds[rowS]) * D_ + halfS * 16;
  const float* wdb  = Wd + (size_t)e * D_ * R_;
  const int rowA = wave * 16 + lc;                            // A-frag row

  for (int kb = 0; kb < D_ / 32; ++kb) {
    {   // stage gathered A tile (128 tokens x 32 k), fp32 -> bf16 pairs
      const float4* s = (const float4*)(xrow + kb * 32);
#pragma unroll
      for (int j = 0; j < 4; ++j) {
        float4 f = s[j];
        sm.p1.a[rowS][halfS * 8 + 2 * j]     = packbf(f.x, f.y);
        sm.p1.a[rowS][halfS * 8 + 2 * j + 1] = packbf(f.z, f.w);
      }
    }
    {   // stage Wd tile (32 k x 128 r) transposed: [r][k2]
      int k2 = tid >> 4, cb = (tid & 15) * 8;
      const float* s0 = wdb + (size_t)(kb * 32 + 2 * k2) * R_ + cb;
      const float* s1 = s0 + R_;
#pragma unroll
      for (int j = 0; j < 8; ++j) sm.p1.bwT[cb + j][k2] = packbf(s0[j], s1[j]);
    }
    __syncthreads();
    Frag af;   // A 16x32 bf16 fragment: two contiguous 16B quads
    af.q[0] = *(const uint4*)&sm.p1.a[rowA][lh * 4];
    af.q[1] = *(const uint4*)&sm.p1.a[rowA][8 + lh * 4];
#pragma unroll
    for (int nt = 0; nt < 8; ++nt) {
      Frag bfr;  // B 32x16 bf16 fragment: contiguous in transposed tile
      const unsigned* bp = &sm.p1.bwT[nt * 16 + lc][lh * 4];
      bfr.q[0] = *(const uint4*)(bp);
      bfr.q[1] = *(const uint4*)(bp + 8);
      acc[nt] = __builtin_amdgcn_wmma_f32_16x16x32_bf16(
          false, af.v, false, bfr.v, (short)0, acc[nt], false, false);
    }
    __syncthreads();
  }

  // bias + relu, keep h in LDS as bf16 (row-major so pairs along k are packed)
#pragma unroll
  for (int nt = 0; nt < 8; ++nt) {
#pragma unroll
    for (int v = 0; v < 8; ++v) {
      int r = wave * 16 + v + lh * 8;          // C layout: M = vgpr + 8*half
      int c = nt * 16 + lc;
      float hv = fmaxf(acc[nt][v] + bd[e * R_ + c], 0.0f);
      hLDS[r][c] = (unsigned short)f2bf(hv);
    }
  }
  __syncthreads();

  // ----- Phase 2: up = h[128 x R] @ Wu[R x D] + bu; gated scatter-add -----
  const unsigned (*hU)[68] = (const unsigned (*)[68])hLDS;
  Frag hF[4];                                   // all A-fragments of h (K=128)
#pragma unroll
  for (int kc = 0; kc < 4; ++kc) {
    hF[kc].q[0] = *(const uint4*)&hU[rowA][kc * 16 + lh * 4];
    hF[kc].q[1] = *(const uint4*)&hU[rowA][kc * 16 + 8 + lh * 4];
  }

  const float* wub = Wu + (size_t)e * R_ * D_;
  const int k2S = tid >> 2, cbS = (tid & 3) * 16;

  for (int n0 = 0; n0 < 32; ++n0) {
    {   // stage Wu tile (128 k x 64 d) transposed: [d][k2]
      const float* s0 = wub + (size_t)(2 * k2S) * D_ + n0 * 64 + cbS;
      const float* s1 = s0 + D_;
      if (n0 + 1 < 32) __builtin_prefetch(s0 + 64, 0, 1);   // next tile
#pragma unroll
      for (int j = 0; j < 16; ++j) sm.b2T[cbS + j][k2S] = packbf(s0[j], s1[j]);
    }
    __syncthreads();
    v8f a2[4];
#pragma unroll
    for (int i = 0; i < 4; ++i) a2[i] = zf;
#pragma unroll
    for (int nt = 0; nt < 4; ++nt) {
#pragma unroll
      for (int kc = 0; kc < 4; ++kc) {
        Frag bfr;
        const unsigned* bp = &sm.b2T[nt * 16 + lc][kc * 16 + lh * 4];
        bfr.q[0] = *(const uint4*)(bp);
        bfr.q[1] = *(const uint4*)(bp + 8);
        a2[nt] = __builtin_amdgcn_wmma_f32_16x16x32_bf16(
            false, hF[kc].v, false, bfr.v, (short)0, a2[nt], false, false);
      }
    }
    // epilogue: out[b, tok, d] += gate * (up + bu + x_in)
#pragma unroll
    for (int nt = 0; nt < 4; ++nt) {
#pragma unroll
      for (int v = 0; v < 8; ++v) {
        int r   = wave * 16 + v + lh * 8;
        int tok = tokLds[r];
        int d   = n0 * 64 + nt * 16 + lc;
        float xin = x[((size_t)(b * N_) + tok) * D_ + d];
        float val = (a2[nt][v] + bu[(size_t)e * D_ + d] + xin) * gateLds[r];
        atomicAdd(out + ((size_t)(b * N_) + tok) * D_ + d, val);
      }
    }
    __syncthreads();
  }
}

// ---------------------------------------------------------------------------
extern "C" void kernel_launch(void* const* d_in, const int* in_sizes, int n_in,
                              void* d_out, int out_size, void* d_ws, size_t ws_size,
                              hipStream_t stream) {
  (void)in_sizes; (void)n_in; (void)out_size; (void)ws_size;
  const float* x        = (const float*)d_in[0];
  const float* residual = (const float*)d_in[1];
  const float* Wg       = (const float*)d_in[2];
  const float* bg       = (const float*)d_in[3];
  const float* Wd       = (const float*)d_in[4];
  const float* bd       = (const float*)d_in[5];
  const float* Wu       = (const float*)d_in[6];
  const float* bu       = (const float*)d_in[7];
  float* out = (float*)d_out;

  float* logits = (float*)d_ws;                                        // B*N*E
  int*   idx    = (int*)((char*)d_ws + (size_t)B_ * N_ * E_ * 4);      // B*E*K
  float* gates  = (float*)((char*)d_ws + (size_t)B_ * N_ * E_ * 4
                                       + (size_t)B_ * E_ * K_ * 4);    // B*E*K

  // combine base = residual (scatter-adds land on top)
  hipMemcpyAsync(out, residual, (size_t)B_ * N_ * D_ * sizeof(float),
                 hipMemcpyDeviceToDevice, stream);

  moa_gate_kernel<<<dim3(B_ * N_ / 8), 256, 0, stream>>>(x, Wg, bg, logits);
  moa_topk_kernel<<<dim3(E_, B_), 256, 0, stream>>>(logits, idx, gates);
  moa_adapter_kernel<<<dim3(K_ / 128, E_, B_), 256, 0, stream>>>(
      x, Wd, bd, Wu, bu, idx, gates, out);
}
